// EfficientViMBlock_60730837565760
// MI455X (gfx1250) — matrix operand order repaired and checked by reading, not compiled, not run on hardware
//
#include <hip/hip_runtime.h>
#include <hip/hip_bf16.h>

// ---------------- problem constants ----------------
#define BB   32
#define DIM  256
#define HGT  64
#define WIDE 64
#define LL   4096      // HGT*WIDE
#define SS   64        // state dim
#define DI   256       // d_inner
#define HID  1024
#define O3S  192       // 3*S
#define EPS  1e-5f

typedef __attribute__((ext_vector_type(16))) __bf16 v16bf;
typedef __attribute__((ext_vector_type(8)))  __bf16 v8bf;
typedef __attribute__((ext_vector_type(8)))  float  v8f;

static __device__ __forceinline__ __bf16 to_bf(float x)   { return (__bf16)x; }
static __device__ __forceinline__ __bf16 to_bf(__bf16 x)  { return x; }
static __device__ __forceinline__ float  sigmoidf_(float x){ return 1.0f / (1.0f + __expf(-x)); }

// ---------------------------------------------------------------------------
// LDS fragment fetch (layouts per CDNA5 ISA 7.12.2, wave32)
// As: TM rows (m) x 32 cols (k), k contiguous          -> A frag = 2x 16B runs
// Bs: TN rows (n) x 32 cols (k), k contiguous (n-major)-> B frag = 2x 16B runs
// ---------------------------------------------------------------------------
static __device__ __forceinline__ v16bf lds_frag_a(const __bf16* As, int mbase, int lane)
{
  const __bf16* p = As + (mbase + (lane & 15)) * 32 + ((lane >> 4) << 3);
  v8bf lo = *(const v8bf*)p;          // K = kh .. kh+7
  v8bf hi = *(const v8bf*)(p + 16);   // K = kh+16 .. kh+23
  return __builtin_shufflevector(lo, hi, 0,1,2,3,4,5,6,7,8,9,10,11,12,13,14,15);
}
static __device__ __forceinline__ v16bf lds_frag_b(const __bf16* Bs, int nbase, int lane)
{
  const __bf16* p = Bs + (nbase + (lane & 15)) * 32 + ((lane >> 4) << 3);
  v8bf lo = *(const v8bf*)p;          // K = kh .. kh+7
  v8bf hi = *(const v8bf*)(p + 16);   // K = kh+16 .. kh+23
  return __builtin_shufflevector(lo, hi, 0,1,2,3,4,5,6,7,8,9,10,11,12,13,14,15);
}

// ---------------------------------------------------------------------------
// Generic LDS-tiled WMMA GEMM: D[b, m, n] = epi( sum_k A[m,k]*B[k,n] )
//   A element: Aall[b*batchA + m*SAM + k]          (k always contiguous)
//   B element: Ball[b*batchB + k*SBK + n*SBN]      (SBN==1: row-major;
//                                                   else SBK==1: k-contiguous)
// Wave grid: TMW x TNW waves, each computing a 32x32 quadrant of the
// (TM = 32*TMW) x (TN = 32*TNW) block tile. TNW must be 2 here.
// EPI: 0 = store f32 | 1 = alpha-blend(aux) | 2 = BN+ReLU -> TD | 3 = BN+blend
// ---------------------------------------------------------------------------
template <int EPI, int TMW, int TNW, long SAM, long SBK, long SBN,
          typename TB, typename TD>
__global__ __launch_bounds__(TMW * TNW * 32)
void k_gemm_tiled(const float* __restrict__ Aall, long batchA,
                  const TB* __restrict__ Ball, long batchB,
                  TD* __restrict__ Dall, long batchD, long ldD,
                  int M, int N, int K,
                  const float* __restrict__ p0, const float* __restrict__ p1,
                  const float* __restrict__ p2, const float* __restrict__ p3,
                  const float* __restrict__ p4,
                  const float* __restrict__ aux, long batchAux)
{
  constexpr int TM = TMW * 32;
  constexpr int TN = TNW * 32;
  constexpr int NT = TMW * TNW * 32;         // threads per block
  constexpr int CHA = (TM * 32) / NT;        // 16 (TNW==2)
  constexpr int CHB = (TN * 32) / NT;        // 16 (TMW==2) or 4 (TMW==8)

  const int nb  = N / TN;
  const int mb  = M / TM;
  const int bid = blockIdx.x;
  const int per = mb * nb;
  const int b   = bid / per;
  const int r   = bid % per;
  const int m0  = (r / nb) * TM;
  const int n0  = (r % nb) * TN;

  const float* Ab = Aall + (long)b * batchA;
  const TB*    Bb = Ball + (long)b * batchB;

  __shared__ __bf16 As[TM * 32];
  __shared__ __bf16 Bs[TN * 32];

  const int tid  = threadIdx.x;
  const int lane = tid & 31;
  const int wave = tid >> 5;
  const int wy   = (wave / TNW) << 5;
  const int wx   = (wave % TNW) << 5;

  v8f acc00 = {}, acc01 = {}, acc10 = {}, acc11 = {};

  for (int k0 = 0; k0 < K; k0 += 32) {
    __syncthreads();
    // ---- stage A tile: CHA contiguous elements per thread ----
    {
      const int idx = tid * CHA;
      const int m   = idx >> 5;
      const int ko  = idx & 31;
      const float* src = Ab + (long)(m0 + m) * SAM + (k0 + ko);
      if (k0 + 32 < K) __builtin_prefetch(src + 32, 0, 1);
      __bf16* dst = As + m * 32 + ko;
#pragma unroll
      for (int v = 0; v < CHA / 8; ++v) {
        v8bf pk;
#pragma unroll
        for (int i = 0; i < 8; ++i) pk[i] = to_bf(src[v * 8 + i]);
        *(v8bf*)(dst + v * 8) = pk;
      }
    }
    // ---- stage B tile into n-major layout ----
    if constexpr (SBN == 1) {
      // B row-major (k,n): thread -> (k = tid&31, CHB consecutive n)
      const int k  = tid & 31;
      const int ng = (tid >> 5) * CHB;
      const TB* src = Bb + (long)(k0 + k) * SBK + (n0 + ng);
      if (k0 + 32 < K) __builtin_prefetch(src + 32 * SBK, 0, 1);
#pragma unroll
      for (int i = 0; i < CHB; ++i)
        Bs[(ng + i) * 32 + k] = to_bf(src[i]);
    } else {
      // B k-contiguous (SBK==1): CHB contiguous elements per thread
      const int idx = tid * CHB;
      const int n   = idx >> 5;
      const int ko  = idx & 31;
      const TB* src = Bb + (long)(n0 + n) * SBN + (k0 + ko);
      if (k0 + 32 < K) __builtin_prefetch(src + 32, 0, 1);
      __bf16* dst = Bs + n * 32 + ko;
      if constexpr (CHB >= 8) {
#pragma unroll
        for (int v = 0; v < CHB / 8; ++v) {
          v8bf pk;
#pragma unroll
          for (int i = 0; i < 8; ++i) pk[i] = to_bf(src[v * 8 + i]);
          *(v8bf*)(dst + v * 8) = pk;
        }
      } else {
#pragma unroll
        for (int i = 0; i < CHB; ++i) dst[i] = to_bf(src[i]);
      }
    }
    __syncthreads();

    // ---- 4 WMMAs per K-step, reusing 2 A + 2 B fragments ----
    const v16bf a0 = lds_frag_a(As, wy,      lane);
    const v16bf a1 = lds_frag_a(As, wy + 16, lane);
    const v16bf b0 = lds_frag_b(Bs, wx,      lane);
    const v16bf b1 = lds_frag_b(Bs, wx + 16, lane);
    acc00 = __builtin_amdgcn_wmma_f32_16x16x32_bf16(false, a0, false, b0, (short)0, acc00, false, false);
    acc01 = __builtin_amdgcn_wmma_f32_16x16x32_bf16(false, a0, false, b1, (short)0, acc01, false, false);
    acc10 = __builtin_amdgcn_wmma_f32_16x16x32_bf16(false, a1, false, b0, (short)0, acc10, false, false);
    acc11 = __builtin_amdgcn_wmma_f32_16x16x32_bf16(false, a1, false, b1, (short)0, acc11, false, false);
  }

  // ---- epilogue ----
  TD* Db = Dall + (long)b * batchD;
  const float* auxb = (EPI == 1 || EPI == 3) ? (aux + (long)b * batchAux) : nullptr;
#pragma unroll
  for (int wi = 0; wi < 2; ++wi) {
#pragma unroll
    for (int wj = 0; wj < 2; ++wj) {
      const v8f& a = wi ? (wj ? acc11 : acc10) : (wj ? acc01 : acc00);
#pragma unroll
      for (int r8 = 0; r8 < 8; ++r8) {
        const int m = m0 + wy + wi * 16 + r8 + ((lane >> 4) << 3);
        const int n = n0 + wx + wj * 16 + (lane & 15);
        const float v = a[r8];
        const long di = (long)m * ldD + n;
        if constexpr (EPI == 0) {
          Db[di] = (TD)v;
        } else if constexpr (EPI == 1) {
          const float al = sigmoidf_(p0[m]);
          Db[di] = (TD)((1.f - al) * auxb[di] + al * v);
        } else if constexpr (EPI == 2) {
          const float bn = (v - p2[m]) * p0[m] * rsqrtf(p3[m] + EPS) + p1[m];
          Db[di] = (TD)fmaxf(bn, 0.f);
        } else {
          const float bn = (v - p2[m]) * p0[m] * rsqrtf(p3[m] + EPS) + p1[m];
          const float al = sigmoidf_(p4[m]);
          Db[di] = (TD)((1.f - al) * auxb[di] + al * bn);
        }
      }
    }
  }
}

// ---------------------------------------------------------------------------
// depthwise 3x3 conv + BN + sigmoid(alpha) residual blend (fp32)
// optional bf16 side copy of the result (for bf16 GEMM operand reuse)
// ---------------------------------------------------------------------------
__global__ void k_dwconv_bn_blend(const float* __restrict__ in,
                                  const float* __restrict__ w9,
                                  const float* __restrict__ g,
                                  const float* __restrict__ bb,
                                  const float* __restrict__ mn,
                                  const float* __restrict__ vv,
                                  const float* __restrict__ alpha_row,
                                  float* __restrict__ out,
                                  __bf16* __restrict__ out_bf)
{
  const long idx = (long)blockIdx.x * blockDim.x + threadIdx.x;
  const long total = (long)BB * DIM * LL;
  if (idx >= total) return;
  const int x = (int)(idx & (WIDE - 1));
  const int y = (int)((idx >> 6) & (HGT - 1));
  const int c = (int)((idx >> 12) & (DIM - 1));
  const long b = idx >> 20;

  const float* img = in + (b * DIM + c) * (long)LL;
  float s = 0.f;
#pragma unroll
  for (int dy = -1; dy <= 1; ++dy)
#pragma unroll
    for (int dx = -1; dx <= 1; ++dx) {
      const int yy = y + dy, xx = x + dx;
      if (yy >= 0 && yy < HGT && xx >= 0 && xx < WIDE)
        s += img[yy * WIDE + xx] * w9[c * 9 + (dy + 1) * 3 + (dx + 1)];
    }
  const float bn = (s - mn[c]) * g[c] * rsqrtf(vv[c] + EPS) + bb[c];
  const float a  = sigmoidf_(alpha_row[c]);
  const float r  = (1.f - a) * in[idx] + a * bn;
  out[idx] = r;
  if (out_bf) out_bf[idx] = to_bf(r);
}

// plain depthwise 3x3 (mixer path, C = 192)
__global__ void k_dwconv_plain(const float* __restrict__ in,
                               const float* __restrict__ w9,
                               float* __restrict__ out)
{
  const long idx = (long)blockIdx.x * blockDim.x + threadIdx.x;
  const long total = (long)BB * O3S * LL;
  if (idx >= total) return;
  const int x = (int)(idx & (WIDE - 1));
  const int y = (int)((idx >> 6) & (HGT - 1));
  const int c = (int)((idx >> 12) % O3S);
  const long b = idx / ((long)O3S * LL);

  const float* img = in + (b * O3S + c) * (long)LL;
  float s = 0.f;
#pragma unroll
  for (int dy = -1; dy <= 1; ++dy)
#pragma unroll
    for (int dx = -1; dx <= 1; ++dx) {
      const int yy = y + dy, xx = x + dx;
      if (yy >= 0 && yy < HGT && xx >= 0 && xx < WIDE)
        s += img[yy * WIDE + xx] * w9[c * 9 + (dy + 1) * 3 + (dx + 1)];
    }
  out[idx] = s;
}

// ---------------------------------------------------------------------------
// LayerNorm over channel dim (per (b,l))
// ---------------------------------------------------------------------------
__global__ void k_layernorm(const float* __restrict__ x,
                            const float* __restrict__ lw,
                            const float* __restrict__ lb,
                            float* __restrict__ xn)
{
  const long idx = (long)blockIdx.x * blockDim.x + threadIdx.x;
  if (idx >= (long)BB * LL) return;
  const long b = idx >> 12;
  const int  l = (int)(idx & (LL - 1));
  const float* px = x + b * DIM * (long)LL + l;
  float s1 = 0.f, s2 = 0.f;
  for (int c = 0; c < DIM; ++c) {
    const float v = px[(long)c * LL];
    s1 += v; s2 += v * v;
  }
  const float mean = s1 * (1.f / DIM);
  const float var  = s2 * (1.f / DIM) - mean * mean;
  const float inv  = rsqrtf(var + EPS);
  float* po = xn + b * DIM * (long)LL + l;
  for (int c = 0; c < DIM; ++c)
    po[(long)c * LL] = (px[(long)c * LL] - mean) * inv * lw[c] + lb[c];
}

// ---------------------------------------------------------------------------
// softmax over L of (dt + A) times Bm  ->  AB   (block per (b,s))
// ---------------------------------------------------------------------------
__global__ void k_softmax_ab(const float* __restrict__ bcdt,
                             const float* __restrict__ A_param,
                             float* __restrict__ AB)
{
  const int bs = blockIdx.x;
  const int b  = bs >> 6;
  const int s  = bs & (SS - 1);
  const float* dt = bcdt + ((long)b * O3S + 2 * SS + s) * LL;
  const float* Bm = bcdt + ((long)b * O3S + s) * LL;
  const float  ap = A_param[s];
  __shared__ float red[256];
  const int tid = threadIdx.x;

  float mx = -3.4e38f;
  for (int l = tid; l < LL; l += 256) mx = fmaxf(mx, dt[l] + ap);
  red[tid] = mx; __syncthreads();
  for (int o = 128; o > 0; o >>= 1) {
    if (tid < o) red[tid] = fmaxf(red[tid], red[tid + o]);
    __syncthreads();
  }
  mx = red[0]; __syncthreads();

  float sm = 0.f;
  for (int l = tid; l < LL; l += 256) sm += __expf(dt[l] + ap - mx);
  red[tid] = sm; __syncthreads();
  for (int o = 128; o > 0; o >>= 1) {
    if (tid < o) red[tid] += red[tid + o];
    __syncthreads();
  }
  const float inv = 1.f / red[0];

  float* out = AB + ((long)b * SS + s) * LL;
  for (int l = tid; l < LL; l += 256)
    out[l] = __expf(dt[l] + ap - mx) * inv * Bm[l];
}

// ---------------------------------------------------------------------------
// hz = hz_w @ h ; gate: hg = hh*silu(z) + hh*D      (small, scalar)
// ---------------------------------------------------------------------------
__global__ void k_hz_gate(const float* __restrict__ hz_w,
                          const float* __restrict__ h,
                          const float* __restrict__ Dp,
                          float* __restrict__ hg)
{
  const long idx = (long)blockIdx.x * blockDim.x + threadIdx.x;
  if (idx >= (long)BB * DI * SS) return;
  const int  s = (int)(idx & (SS - 1));
  const int  o = (int)((idx >> 6) & (DI - 1));
  const long b = idx >> 14;
  const float* hb = h + b * DIM * (long)SS + s;
  const float* w1 = hz_w + (long)o * DIM;
  const float* w2 = hz_w + (long)(DI + o) * DIM;
  float hh = 0.f, z = 0.f;
  for (int c = 0; c < DIM; ++c) {
    const float v = hb[(long)c * SS];
    hh += w1[c] * v;
    z  += w2[c] * v;
  }
  const float sil = z * sigmoidf_(z);
  hg[idx] = hh * sil + hh * Dp[0];
}

// hout = out_w @ hg  (written straight into d_out tail)
__global__ void k_out_proj(const float* __restrict__ out_w,
                           const float* __restrict__ hg,
                           float* __restrict__ hout)
{
  const long idx = (long)blockIdx.x * blockDim.x + threadIdx.x;
  if (idx >= (long)BB * DIM * SS) return;
  const int  s = (int)(idx & (SS - 1));
  const int  o = (int)((idx >> 6) & (DIM - 1));
  const long b = idx >> 14;
  const float* hb = hg + b * DI * (long)SS + s;
  const float* w  = out_w + (long)o * DI;
  float acc = 0.f;
  for (int c = 0; c < DI; ++c) acc += w[c] * hb[(long)c * SS];
  hout[idx] = acc;
}

// ---------------------------------------------------------------------------
// host side
// ---------------------------------------------------------------------------
extern "C" void kernel_launch(void* const* d_in, const int* in_sizes, int n_in,
                              void* d_out, int out_size, void* d_ws, size_t ws_size,
                              hipStream_t stream)
{
  (void)in_sizes; (void)n_in; (void)out_size; (void)ws_size;
  const float* x_in   = (const float*)d_in[0];
  const float* dw1_w  = (const float*)d_in[1];
  const float* dw1_g  = (const float*)d_in[2];
  const float* dw1_b  = (const float*)d_in[3];
  const float* dw1_m  = (const float*)d_in[4];
  const float* dw1_v  = (const float*)d_in[5];
  const float* ln_w   = (const float*)d_in[6];
  const float* ln_b   = (const float*)d_in[7];
  const float* bcdt_w = (const float*)d_in[8];
  const float* dwm_w  = (const float*)d_in[9];
  const float* A_par  = (const float*)d_in[10];
  const float* hz_w   = (const float*)d_in[11];
  const float* out_w  = (const float*)d_in[12];
  const float* D_par  = (const float*)d_in[13];
  const float* dw2_w  = (const float*)d_in[14];
  const float* dw2_g  = (const float*)d_in[15];
  const float* dw2_b  = (const float*)d_in[16];
  const float* dw2_m  = (const float*)d_in[17];
  const float* dw2_v  = (const float*)d_in[18];
  const float* fc1_w  = (const float*)d_in[19];
  const float* fc1_g  = (const float*)d_in[20];
  const float* fc1_b  = (const float*)d_in[21];
  const float* fc1_m  = (const float*)d_in[22];
  const float* fc1_v  = (const float*)d_in[23];
  const float* fc2_w  = (const float*)d_in[24];
  const float* fc2_g  = (const float*)d_in[25];
  const float* fc2_b  = (const float*)d_in[26];
  const float* fc2_m  = (const float*)d_in[27];
  const float* fc2_v  = (const float*)d_in[28];
  const float* alpha  = (const float*)d_in[29];

  // workspace arena (floats), x2 overlays xn (dead after h-GEMM)
  const long SZ_X    = (long)BB * DIM * LL;
  const long SZ_BCDT = (long)BB * O3S * LL;
  const long SZ_S    = (long)BB * SS * LL;
  const long SZ_H    = (long)BB * DIM * SS;
  float* ws      = (float*)d_ws;
  float* x1      = ws;
  float* xn      = ws + SZ_X;
  float* bcdt_a  = ws + 2 * SZ_X;
  float* bcdt_b  = bcdt_a + SZ_BCDT;
  float* AB      = bcdt_b + SZ_BCDT;
  float* h       = AB + SZ_S;
  float* hg      = h + SZ_H;
  float* x3      = hg + SZ_H;
  __bf16* t      = (__bf16*)(x3 + SZ_X);           // BB*HID*LL bf16
  __bf16* x3h    = t + (long)BB * HID * LL;        // BB*DIM*LL bf16
  float* x2      = xn;  // reuse

  float* out_x    = (float*)d_out;
  float* out_hout = out_x + SZ_X;

  const int TPB = 256;
  const long nElem = SZ_X;
  const float* nul = nullptr;

  // 1) dwconv1 + BN + alpha0 blend
  k_dwconv_bn_blend<<<(unsigned)((nElem + TPB - 1) / TPB), TPB, 0, stream>>>(
      x_in, dw1_w, dw1_g, dw1_b, dw1_m, dw1_v, alpha + 0 * DIM, x1, nullptr);

  // 2) LayerNorm
  k_layernorm<<<(unsigned)(((long)BB * LL + TPB - 1) / TPB), TPB, 0, stream>>>(
      x1, ln_w, ln_b, xn);

  // 3) bc_dt = bcdt_w(192x256) @ xn(256xL)        [64x64 tiles]
  k_gemm_tiled<0, 2, 2, (long)DIM, (long)LL, 1L, float, float>
      <<<BB * (O3S / 64) * (LL / 64), 128, 0, stream>>>(
      bcdt_w, 0L, xn, (long)DIM * LL, bcdt_a, (long)O3S * LL, (long)LL,
      O3S, LL, DIM, nul, nul, nul, nul, nul, nul, 0L);

  // 4) mixer depthwise conv
  k_dwconv_plain<<<(unsigned)(((long)BB * O3S * LL + TPB - 1) / TPB), TPB, 0,
                   stream>>>(bcdt_a, dwm_w, bcdt_b);

  // 5) softmax * Bm -> AB
  k_softmax_ab<<<BB * SS, 256, 0, stream>>>(bcdt_b, A_par, AB);

  // 6) h = xn(256xL) @ AB^T(Lx64)                 [64x64 tiles, B k-contig]
  k_gemm_tiled<0, 2, 2, (long)LL, 1L, (long)LL, float, float>
      <<<BB * (DIM / 64) * (SS / 64), 128, 0, stream>>>(
      xn, (long)DIM * LL, AB, (long)SS * LL, h, (long)DIM * SS, (long)SS,
      DIM, SS, LL, nul, nul, nul, nul, nul, nul, 0L);

  // 7) hz/gate + out projection (hout -> d_out tail)
  k_hz_gate<<<(unsigned)(((long)BB * DI * SS + TPB - 1) / TPB), TPB, 0, stream>>>(
      hz_w, h, D_par, hg);
  k_out_proj<<<(unsigned)(((long)BB * DIM * SS + TPB - 1) / TPB), TPB, 0, stream>>>(
      out_w, hg, out_hout);

  // 8) y = hout(256x64) @ Cm(64xL), fused alpha1 blend  [256x64 tiles]
  k_gemm_tiled<1, 8, 2, (long)SS, (long)LL, 1L, float, float>
      <<<BB * (DIM / 256) * (LL / 64), 512, 0, stream>>>(
      out_hout, (long)DIM * SS, bcdt_b + (long)SS * LL, (long)O3S * LL,
      x2, (long)DIM * LL, (long)LL,
      DIM, LL, SS, alpha + 1 * DIM, nul, nul, nul, nul, x1, (long)DIM * LL);

  // 9) dwconv2 + BN + alpha2 blend -> x3 (+ bf16 copy x3h)
  k_dwconv_bn_blend<<<(unsigned)((nElem + TPB - 1) / TPB), TPB, 0, stream>>>(
      x2, dw2_w, dw2_g, dw2_b, dw2_m, dw2_v, alpha + 2 * DIM, x3, x3h);

  // 10) t = relu(BN(fc1_w(1024x256) @ x3h(256xL))) -> bf16   [256x64 tiles]
  k_gemm_tiled<2, 8, 2, (long)DIM, (long)LL, 1L, __bf16, __bf16>
      <<<BB * (HID / 256) * (LL / 64), 512, 0, stream>>>(
      fc1_w, 0L, x3h, (long)DIM * LL, t, (long)HID * LL, (long)LL,
      HID, LL, DIM, fc1_g, fc1_b, fc1_m, fc1_v, nul, nul, 0L);

  // 11) out = (1-a3)*x3 + a3*BN(fc2_w(256x1024) @ t(1024xL)) [256x64 tiles]
  k_gemm_tiled<3, 8, 2, (long)HID, (long)LL, 1L, __bf16, float>
      <<<BB * (DIM / 256) * (LL / 64), 512, 0, stream>>>(
      fc2_w, 0L, t, (long)HID * LL, out_x, (long)DIM * LL, (long)LL,
      DIM, LL, HID, fc2_g, fc2_b, fc2_m, fc2_v, alpha + 3 * DIM,
      x3, (long)DIM * LL);
}
